// EncoderBlock_8272107012274
// MI455X (gfx1250) — compile-verified, matrix operand order checked by reading, not run on hardware
//
#include <hip/hip_runtime.h>

// ---------------------------------------------------------------------------
// CDNA5 (gfx1250) feature detection: async global->LDS copies (ASYNCcnt)
// ---------------------------------------------------------------------------
#if defined(__has_builtin)
#if __has_builtin(__builtin_amdgcn_global_load_async_to_lds_b128) && \
    __has_builtin(__builtin_amdgcn_s_wait_asynccnt)
#define USE_ASYNC_LDS 1
#endif
#endif

typedef __bf16 bf16_t;
typedef __attribute__((ext_vector_type(16))) __bf16 v16bf;
typedef __attribute__((ext_vector_type(8)))  float  v8f;
typedef __attribute__((ext_vector_type(4)))  int    v4i;

constexpr int Bc = 4, Sc = 2048, Dc = 1024, Hc = 16, DKc = 64, DFFc = 4096;
constexpr int Mrows = Bc * Sc; // 8192
constexpr float EPSc = 1e-6f;

union Frag32B { uint4 u4[2]; v16bf v; bf16_t h[16]; };

__device__ inline v8f vzero8() {
  v8f z;
#pragma unroll
  for (int e = 0; e < 8; ++e) z[e] = 0.f;
  return z;
}

__device__ inline v8f wmma_bf16(v16bf a, v16bf b, v8f c) {
  return __builtin_amdgcn_wmma_f32_16x16x32_bf16(
      false, a, false, b, (short)0, c, false, false);
}

// One 16-byte global->LDS copy per lane. Async on gfx1250 toolchains that
// expose the builtin (tracked by ASYNCcnt), plain VGPR round-trip otherwise.
// This toolchain's builtin takes generic v4i* pointers (per its diagnostic).
__device__ inline void cp16(bf16_t* lds, const bf16_t* g) {
#if defined(USE_ASYNC_LDS)
  __builtin_amdgcn_global_load_async_to_lds_b128((v4i*)g, (v4i*)lds, 0, 0);
#else
  *(uint4*)lds = *(const uint4*)g;
#endif
}
__device__ inline void cp_wait() {
#if defined(USE_ASYNC_LDS)
  __builtin_amdgcn_s_wait_asynccnt(0);
#endif
}

// Load one 16x32 bf16 fragment in the ISA-documented A/B VGPR layout:
//   lane L (0-15):  row = L,    bytes [0,16) and [32,48)  (K 0..7, 16..23)
//   lane L (16-31): row = L-16, bytes [16,32) and [48,64) (K 8..15, 24..31)
__device__ inline v16bf frag_load(const bf16_t* tile, int lane, int row_stride) {
  Frag32B f;
  const char* p = (const char*)(tile + (size_t)(lane & 15) * row_stride)
                + ((lane >> 4) & 1) * 16;
  f.u4[0] = *(const uint4*)(p);
  f.u4[1] = *(const uint4*)(p + 32);
  return f.v;
}

// ---------------------------------------------------------------------------
// fp32 [K][N] -> bf16 [N][K] tiled transpose (weights, once per launch)
// ---------------------------------------------------------------------------
__global__ __launch_bounds__(256)
void cvt_transpose_bf16_kernel(const float* __restrict__ src,  // [K][N]
                               bf16_t* __restrict__ dst,       // [N][K]
                               int K, int N) {
  __shared__ float T[32][33];
  int tileN = blockIdx.x * 32;
  int tileK = blockIdx.y * 32;
  int tx = threadIdx.x & 31, ty = threadIdx.x >> 5;  // ty 0..7
#pragma unroll
  for (int i = 0; i < 4; ++i) {
    int kk = ty + i * 8;
    T[kk][tx] = src[(size_t)(tileK + kk) * N + tileN + tx];
  }
  __syncthreads();
#pragma unroll
  for (int i = 0; i < 4; ++i) {
    int nn = ty + i * 8;
    dst[(size_t)(tileN + nn) * K + tileK + tx] = (bf16_t)T[tx][nn];
  }
}

// ---------------------------------------------------------------------------
// LayerNorm over last dim (D=1024), scalar alpha/beta, bf16 output.
// ---------------------------------------------------------------------------
__global__ __launch_bounds__(256)
void layernorm_bf16_kernel(const float* __restrict__ x,
                           const float* __restrict__ alpha,
                           const float* __restrict__ beta,
                           bf16_t* __restrict__ out) {
  __shared__ float ps[8], ps2[8];
  int row = blockIdx.x;
  const float* xr = x + (size_t)row * Dc;
  float vals[4];
  float s = 0.f, s2 = 0.f;
#pragma unroll
  for (int i = 0; i < 4; ++i) {
    float v = xr[threadIdx.x + i * 256];
    vals[i] = v; s += v; s2 += v * v;
  }
#pragma unroll
  for (int m = 16; m >= 1; m >>= 1) {
    s  += __shfl_xor(s, m, 32);
    s2 += __shfl_xor(s2, m, 32);
  }
  int wave = threadIdx.x >> 5, lane = threadIdx.x & 31;
  if (lane == 0) { ps[wave] = s; ps2[wave] = s2; }
  __syncthreads();
  if (wave == 0) {
    float a = (lane < 8) ? ps[lane]  : 0.f;
    float b = (lane < 8) ? ps2[lane] : 0.f;
#pragma unroll
    for (int m = 4; m >= 1; m >>= 1) {
      a += __shfl_xor(a, m, 32);
      b += __shfl_xor(b, m, 32);
    }
    if (lane == 0) { ps[0] = a; ps2[0] = b; }
  }
  __syncthreads();
  float mean = ps[0] * (1.f / Dc);
  float var  = ps2[0] * (1.f / Dc) - mean * mean;
  float rstd = rsqrtf(var + EPSc);
  float al = alpha[0], be = beta[0];
  bf16_t* orow = out + (size_t)row * Dc;
#pragma unroll
  for (int i = 0; i < 4; ++i)
    orow[threadIdx.x + i * 256] = (bf16_t)(al * (vals[i] - mean) * rstd + be);
}

// ---------------------------------------------------------------------------
// Tiled WMMA GEMM: C[M,N] = A[M,K](bf16) @ W[K,N] + bias, W given pre-
// transposed as WT[N][K] so every LDS stage is a contiguous b128 copy.
// Block tile 128x128x32; 8 waves (4Mx2N); double-buffered async staging.
//   DUAL:     also A@W3+b3, epilogue out = bf16((A@W1+b1) * silu(A@W3+b3))
//   RESID:    fp32 out = acc + bias + resid
//   OUT_BF16: bf16 out
//   OUT_VT:   bf16 out scattered to [B][H][DK][S] (V for attention)
// ---------------------------------------------------------------------------
constexpr int BM = 128, BN = 128, BK = 32, LDT = 48;  // 96B rows, 16B aligned

template <bool OUT_BF16, bool RESID, bool DUAL, bool OUT_VT>
__global__ __launch_bounds__(256)
void gemm_wmma_kernel(const bf16_t* __restrict__ A,    // [M][K]
                      const bf16_t* __restrict__ WT,   // [N][K]
                      const float* __restrict__ bias,
                      const bf16_t* __restrict__ W3T,  // [N][K] (DUAL)
                      const float* __restrict__ bias3,
                      const float* __restrict__ resid,
                      float* __restrict__ outF, bf16_t* __restrict__ outB,
                      int M, int N, int K) {
  constexpr int NBUF = DUAL ? 1 : 2;
  __shared__ __align__(16) bf16_t At[NBUF][BM * LDT];               // 12/24 KB
  __shared__ __align__(16) bf16_t Wt[NBUF][BN * LDT];               // 12/24 KB
  __shared__ __align__(16) bf16_t Wt3[DUAL ? (BN * LDT) : 16];      // 12 KB if DUAL

  int tid = threadIdx.x;
  int wave = tid >> 5, lane = tid & 31;
  int waveM = wave & 3, waveN = wave >> 2;
  int rowBase = blockIdx.y * BM;
  int colBase = blockIdx.x * BN;
  int halfsel = (lane >> 4) & 1;

  v8f acc[2][4];
  v8f acc3[DUAL ? 2 : 1][DUAL ? 4 : 1];
#pragma unroll
  for (int i = 0; i < 2; ++i)
#pragma unroll
    for (int j = 0; j < 4; ++j) acc[i][j] = vzero8();
  if constexpr (DUAL) {
#pragma unroll
    for (int i = 0; i < 2; ++i)
#pragma unroll
      for (int j = 0; j < 4; ++j) acc3[i][j] = vzero8();
  }

  int ldRow = tid >> 1;            // 0..127
  int ldKh  = (tid & 1) * 16;      // 0 or 16
  const bf16_t* Ag  = A  + (size_t)(rowBase + ldRow) * K + ldKh;
  const bf16_t* Wg  = WT + (size_t)(colBase + ldRow) * K + ldKh;
  const bf16_t* W3g = DUAL ? (W3T + (size_t)(colBase + ldRow) * K + ldKh) : nullptr;
  int ldsOff = ldRow * LDT + ldKh;

  auto stage = [&](int buf, int kb) {
    cp16(&At[buf][ldsOff],     Ag + kb);
    cp16(&At[buf][ldsOff + 8], Ag + kb + 8);
    cp16(&Wt[buf][ldsOff],     Wg + kb);
    cp16(&Wt[buf][ldsOff + 8], Wg + kb + 8);
    if constexpr (DUAL) {
      cp16(&Wt3[ldsOff],     W3g + kb);
      cp16(&Wt3[ldsOff + 8], W3g + kb + 8);
    }
  };

  auto compute = [&](int buf) {
    v16bf afr0 = frag_load(&At[buf][(waveM * 32 +  0) * LDT], lane, LDT);
    v16bf afr1 = frag_load(&At[buf][(waveM * 32 + 16) * LDT], lane, LDT);
#pragma unroll
    for (int nt = 0; nt < 4; ++nt) {
      v16bf bfr = frag_load(&Wt[buf][(waveN * 64 + nt * 16) * LDT], lane, LDT);
      acc[0][nt] = wmma_bf16(afr0, bfr, acc[0][nt]);
      acc[1][nt] = wmma_bf16(afr1, bfr, acc[1][nt]);
      if constexpr (DUAL) {
        v16bf bfr3 = frag_load(&Wt3[(waveN * 64 + nt * 16) * LDT], lane, LDT);
        acc3[0][nt] = wmma_bf16(afr0, bfr3, acc3[0][nt]);
        acc3[1][nt] = wmma_bf16(afr1, bfr3, acc3[1][nt]);
      }
    }
  };

  if constexpr (NBUF == 2) stage(0, 0);
  int cur = 0;
  for (int kb = 0; kb < K; kb += BK) {
    if constexpr (NBUF == 1) stage(0, kb);
    cp_wait();
    __syncthreads();
    if constexpr (NBUF == 2) {
      if (kb + BK < K) stage(cur ^ 1, kb + BK);
    }
    compute(cur);
    __syncthreads();
    if constexpr (NBUF == 2) cur ^= 1;
  }

  // epilogue: C layout -> lane = col(N), element e = row (+8 for lanes 16-31)
#pragma unroll
  for (int mt = 0; mt < 2; ++mt) {
#pragma unroll
    for (int nt = 0; nt < 4; ++nt) {
      int col = colBase + waveN * 64 + nt * 16 + (lane & 15);
      float bv  = bias[col];
      float bv3 = 0.f;
      if constexpr (DUAL) bv3 = bias3[col];
#pragma unroll
      for (int e = 0; e < 8; ++e) {
        int row = rowBase + waveM * 32 + mt * 16 + e + halfsel * 8;
        float v = acc[mt][nt][e] + bv;
        if constexpr (DUAL) {
          float g  = acc3[mt][nt][e] + bv3;
          float sg = g * (1.f / (1.f + __expf(-g)));       // silu(gate)
          outB[(size_t)row * N + col] = (bf16_t)(v * sg);  // value*silu(gate)
        } else if constexpr (OUT_VT) {
          // scatter to vT[B][H][DK][S]; consecutive e => consecutive s
          int s  = row & (Sc - 1);
          int bb = row >> 11;            // row / Sc
          int hh = col >> 6;             // col / DK
          int dk = col & (DKc - 1);
          outB[(((size_t)bb * Hc + hh) * DKc + dk) * Sc + s] = (bf16_t)v;
        } else if constexpr (OUT_BF16) {
          outB[(size_t)row * N + col] = (bf16_t)v;
        } else {
          size_t idx = (size_t)row * N + col;
          if constexpr (RESID) v += resid[idx];
          outF[idx] = v;
        }
      }
    }
  }
}

// ---------------------------------------------------------------------------
// Flash attention: grid (S/128, H, B); 8 waves x 16 q-rows each.
// q,k bf16 [B,S,H,DK]; v bf16 pre-transposed [B][H][DK][S]; ctx bf16 [B,S,D].
// Double-buffered async K/V staging; online softmax fp32.
// ---------------------------------------------------------------------------
__global__ __launch_bounds__(256)
void attention_kernel(const bf16_t* __restrict__ q,
                      const bf16_t* __restrict__ k,
                      const bf16_t* __restrict__ vT,
                      const int* __restrict__ mask,
                      bf16_t* __restrict__ ctx) {
  __shared__ __align__(16) bf16_t Kb[2][32 * 64];     // [key][dk]  8 KB
  __shared__ __align__(16) bf16_t Vt[2][64 * 32];     // [dk][key]  8 KB
  __shared__ __align__(16) bf16_t Ps[8][16 * 48];     // per-wave P 12 KB

  int tid = threadIdx.x, wave = tid >> 5, lane = tid & 31;
  int b = blockIdx.z, head = blockIdx.y;
  int qbase = blockIdx.x * 128 + wave * 16;
  int halfsel = (lane >> 4) & 1;
  const float scale = 0.125f;  // 1/sqrt(64)

  const int strideS = Hc * DKc;
  const bf16_t* qrow0 = q + ((size_t)(b * Sc + qbase) * Hc + head) * DKc;
  v16bf q0 = frag_load(qrow0,      lane, strideS);
  v16bf q1 = frag_load(qrow0 + 32, lane, strideS);

  v8f o[4];
#pragma unroll
  for (int t = 0; t < 4; ++t) o[t] = vzero8();
  float mrun[8], lrun[8];
#pragma unroll
  for (int e = 0; e < 8; ++e) { mrun[e] = -1e30f; lrun[e] = 0.f; }

  // staging index maps
  int kkey = tid >> 3;            // 0..31
  int kdk  = (tid & 7) * 8;       // 0..56
  int vdk  = tid >> 2;            // 0..63
  int vko  = (tid & 3) * 8;       // 0..24
  const size_t kbase = ((size_t)b * Sc * Hc + head) * DKc + kdk;
  const size_t vbase = (((size_t)b * Hc + head) * DKc + vdk) * Sc + vko;

  auto stageKV = [&](int buf, int jb) {
    cp16(&Kb[buf][kkey * 64 + kdk], k  + kbase + (size_t)(jb + kkey) * strideS);
    cp16(&Vt[buf][vdk * 32 + vko],  vT + vbase + jb);
  };

  stageKV(0, 0);
  int cur = 0;
  for (int jb = 0; jb < Sc; jb += 32) {
    cp_wait();
    __syncthreads();
    if (jb + 32 < Sc) stageKV(cur ^ 1, jb + 32);

    // scores for 32 keys: 2 sub-tiles of 16, dk reduced over two 32-chunks
    v8f sc[2];
#pragma unroll
    for (int kt = 0; kt < 2; ++kt) {
      v8f c = vzero8();
      v16bf kb0 = frag_load(&Kb[cur][(kt * 16) * 64 +  0], lane, 64);
      v16bf kb1 = frag_load(&Kb[cur][(kt * 16) * 64 + 32], lane, 64);
      c = wmma_bf16(q0, kb0, c);
      c = wmma_bf16(q1, kb1, c);
      int mval = mask[b * Sc + jb + kt * 16 + (lane & 15)];
#pragma unroll
      for (int e = 0; e < 8; ++e) {
        float s = c[e] * scale;
        if (mval == 0) s = -1e9f;
        c[e] = s;
      }
      sc[kt] = c;
    }

    // online softmax (row stats via half-wave reductions; element e <-> row)
#pragma unroll
    for (int e = 0; e < 8; ++e) {
      float rmax = fmaxf(sc[0][e], sc[1][e]);
#pragma unroll
      for (int off = 8; off >= 1; off >>= 1)
        rmax = fmaxf(rmax, __shfl_xor(rmax, off, 16));
      float mnew = fmaxf(mrun[e], rmax);
      float corr = __expf(mrun[e] - mnew);
      float p0 = __expf(sc[0][e] - mnew);
      float p1 = __expf(sc[1][e] - mnew);
      float rs = p0 + p1;
#pragma unroll
      for (int off = 8; off >= 1; off >>= 1) rs += __shfl_xor(rs, off, 16);
      lrun[e] = lrun[e] * corr + rs;
      mrun[e] = mnew;
#pragma unroll
      for (int t = 0; t < 4; ++t) o[t][e] *= corr;
      int m = e + halfsel * 8;     // transpose C-layout -> A-layout via LDS
      Ps[wave][m * 48 +      (lane & 15)] = (bf16_t)p0;
      Ps[wave][m * 48 + 16 + (lane & 15)] = (bf16_t)p1;
    }

    // O += P(16x32) @ V(32x64)
    v16bf pa = frag_load(Ps[wave], lane, 48);
#pragma unroll
    for (int t = 0; t < 4; ++t) {
      v16bf vb = frag_load(&Vt[cur][(t * 16) * 32], lane, 32);
      o[t] = wmma_bf16(pa, vb, o[t]);
    }
    __syncthreads();
    cur ^= 1;
  }

  // normalize and write ctx ([B,S,D], d = head*64 + dk)
#pragma unroll
  for (int e = 0; e < 8; ++e) {
    float rinv = 1.f / lrun[e];
    int m = e + halfsel * 8;
    int row = qbase + m;
#pragma unroll
    for (int t = 0; t < 4; ++t) {
      int dk = t * 16 + (lane & 15);
      ctx[(size_t)(b * Sc + row) * Dc + head * DKc + dk] = (bf16_t)(o[t][e] * rinv);
    }
  }
}

// ---------------------------------------------------------------------------
// Host-side orchestration
// ---------------------------------------------------------------------------
extern "C" void kernel_launch(void* const* d_in, const int* in_sizes, int n_in,
                              void* d_out, int out_size, void* d_ws, size_t ws_size,
                              hipStream_t stream) {
  const float* x     = (const float*)d_in[0];
  const int*   mask  = (const int*)  d_in[1];
  const float* wq    = (const float*)d_in[2];
  const float* bq    = (const float*)d_in[3];
  const float* wk    = (const float*)d_in[4];
  const float* bk    = (const float*)d_in[5];
  const float* wv    = (const float*)d_in[6];
  const float* bv    = (const float*)d_in[7];
  const float* wo    = (const float*)d_in[8];
  const float* bo    = (const float*)d_in[9];
  const float* w1    = (const float*)d_in[10];
  const float* b1    = (const float*)d_in[11];
  const float* w3    = (const float*)d_in[12];
  const float* b3    = (const float*)d_in[13];
  const float* w2    = (const float*)d_in[14];
  const float* b2    = (const float*)d_in[15];
  const float* alpha1 = (const float*)d_in[16];
  const float* beta1  = (const float*)d_in[17];
  const float* alpha2 = (const float*)d_in[18];
  const float* beta2  = (const float*)d_in[19];
  float* out = (float*)d_out;

  char* wp = (char*)d_ws;
  auto carve = [&](size_t bytes) -> char* {
    char* r = wp;
    wp += (bytes + 255) & ~(size_t)255;
    return r;
  };
  const size_t MD2 = (size_t)Mrows * Dc * sizeof(bf16_t);
  bf16_t* h1b  = (bf16_t*)carve(MD2);
  bf16_t* h2b  = (bf16_t*)carve(MD2);
  bf16_t* qb   = (bf16_t*)carve(MD2);
  bf16_t* kb   = (bf16_t*)carve(MD2);
  bf16_t* vtb  = (bf16_t*)carve(MD2);                              // [B][H][DK][S]
  bf16_t* ctxb = (bf16_t*)carve(MD2);
  float*  x1   = (float*) carve((size_t)Mrows * Dc * sizeof(float));
  bf16_t* gb   = (bf16_t*)carve((size_t)Mrows * DFFc * sizeof(bf16_t));
  bf16_t* wqT  = (bf16_t*)carve((size_t)Dc * Dc * sizeof(bf16_t)); // [N][K]
  bf16_t* wkT  = (bf16_t*)carve((size_t)Dc * Dc * sizeof(bf16_t));
  bf16_t* wvT  = (bf16_t*)carve((size_t)Dc * Dc * sizeof(bf16_t));
  bf16_t* woT  = (bf16_t*)carve((size_t)Dc * Dc * sizeof(bf16_t));
  bf16_t* w1T  = (bf16_t*)carve((size_t)Dc * DFFc * sizeof(bf16_t));
  bf16_t* w3T  = (bf16_t*)carve((size_t)Dc * DFFc * sizeof(bf16_t));
  bf16_t* w2T  = (bf16_t*)carve((size_t)DFFc * Dc * sizeof(bf16_t));

  // 1) weight convert + transpose to bf16 [N][K]
  dim3 gDD(Dc / 32, Dc / 32);
  cvt_transpose_bf16_kernel<<<gDD, 256, 0, stream>>>(wq, wqT, Dc, Dc);
  cvt_transpose_bf16_kernel<<<gDD, 256, 0, stream>>>(wk, wkT, Dc, Dc);
  cvt_transpose_bf16_kernel<<<gDD, 256, 0, stream>>>(wv, wvT, Dc, Dc);
  cvt_transpose_bf16_kernel<<<gDD, 256, 0, stream>>>(wo, woT, Dc, Dc);
  dim3 gDF(DFFc / 32, Dc / 32);
  cvt_transpose_bf16_kernel<<<gDF, 256, 0, stream>>>(w1, w1T, Dc, DFFc);
  cvt_transpose_bf16_kernel<<<gDF, 256, 0, stream>>>(w3, w3T, Dc, DFFc);
  dim3 gFD(Dc / 32, DFFc / 32);
  cvt_transpose_bf16_kernel<<<gFD, 256, 0, stream>>>(w2, w2T, DFFc, Dc);

  // 2) LN1
  layernorm_bf16_kernel<<<Mrows, 256, 0, stream>>>(x, alpha1, beta1, h1b);

  // 3) Q/K/V projections (V scattered to [B][H][DK][S])
  dim3 gP(Dc / BN, Mrows / BM);
  gemm_wmma_kernel<true, false, false, false><<<gP, 256, 0, stream>>>(
      h1b, wqT, bq, nullptr, nullptr, nullptr, nullptr, qb, Mrows, Dc, Dc);
  gemm_wmma_kernel<true, false, false, false><<<gP, 256, 0, stream>>>(
      h1b, wkT, bk, nullptr, nullptr, nullptr, nullptr, kb, Mrows, Dc, Dc);
  gemm_wmma_kernel<false, false, false, true><<<gP, 256, 0, stream>>>(
      h1b, wvT, bv, nullptr, nullptr, nullptr, nullptr, vtb, Mrows, Dc, Dc);

  // 4) attention -> ctx
  dim3 gA(Sc / 128, Hc, Bc);
  attention_kernel<<<gA, 256, 0, stream>>>(qb, kb, vtb, mask, ctxb);

  // 5) output projection + residual: x1 = x + ctx@wo + bo
  gemm_wmma_kernel<false, true, false, false><<<gP, 256, 0, stream>>>(
      ctxb, woT, bo, nullptr, nullptr, x, x1, nullptr, Mrows, Dc, Dc);

  // 6) LN2
  layernorm_bf16_kernel<<<Mrows, 256, 0, stream>>>(x1, alpha2, beta2, h2b);

  // 7) fused SwiGLU: g = bf16((h2@w1+b1) * silu(h2@w3+b3))
  dim3 gF(DFFc / BN, Mrows / BM);
  gemm_wmma_kernel<true, false, true, false><<<gF, 256, 0, stream>>>(
      h2b, w1T, b1, w3T, b3, nullptr, nullptr, gb, Mrows, DFFc, Dc);

  // 8) down-projection + residual: out = x1 + g@w2 + b2
  gemm_wmma_kernel<false, true, false, false><<<gP, 256, 0, stream>>>(
      gb, w2T, b2, nullptr, nullptr, x1, out, nullptr, Mrows, Dc, DFFc);
}
